// SvegaSlowConv2d_5248450036551
// MI455X (gfx1250) — matrix operand order repaired and checked by reading, not compiled, not run on hardware
//
#include <hip/hip_runtime.h>

// CDNA5 / gfx1250 implicit-GEMM conv2d: NCHW fp32 in/out, bf16 WMMA compute.
// GEMM view: M=C_out=256, N'=N*H*W=100352, K=C_in*9=1152.
// Block tile 128x128x32, 8 wave32 waves (4x2), each wave: 2x4 grid of
// v_wmma_f32_16x16x32_bf16 fragments (32x64 output per wave).

typedef __attribute__((ext_vector_type(16))) __bf16 v16bf;
typedef __attribute__((ext_vector_type(8)))  __bf16 v8bf;
typedef __attribute__((ext_vector_type(8)))  float  v8f;
typedef __attribute__((ext_vector_type(4)))  float  v4f;

#define IN_C   128
#define OUT_C  256
#define HW     56
#define NPIX   3136      // 56*56
#define KTOT   1152      // IN_C*3*3
#define BM     128
#define BN     128
#define BK     32
#define LDA    40        // BK + 8 bf16 pad -> 80B row stride, 16B aligned, bank-spread
#define LDB    40

union Half16 { v16bf v; v8bf h[2]; };

__global__ __launch_bounds__(256) void conv_igemm_wmma_bf16(
    const float* __restrict__ X,   // [32,128,56,56]
    const float* __restrict__ Wt,  // [256,128,3,3]
    float* __restrict__ Out)       // [32,256,56,56]
{
  __shared__ __bf16 As[BM * LDA]; // [m][k] row-major, k contiguous
  __shared__ __bf16 Bs[BN * LDB]; // [n][k] row-major, k contiguous (im2col, transposed)

  const int t    = threadIdx.x;
  const int lane = t & 31;
  const int l15  = lane & 15;
  const int wave = t >> 5;
  const int wm   = wave & 3;   // 4 waves along M: 32 rows each
  const int wn   = wave >> 2;  // 2 waves along N: 64 cols each

  const int co0 = blockIdx.y * BM;   // output-channel tile base (0 or 128)
  const int p0  = blockIdx.x * BN;   // flat pixel tile base

  // ---- B gather mapping: one pixel per thread, 16 consecutive k per thread
  const int bn   = t & 127;          // column within tile
  const int kc   = t >> 7;           // which 16-wide k chunk (0/1)
  const int p    = p0 + bn;
  const int nb   = p / NPIX;
  const int prem = p - nb * NPIX;
  const int oh   = prem / HW;
  const int ow   = prem - oh * HW;

  // ---- A load mapping: 2 threads per weight row, 16 contiguous floats each
  const int ar = t >> 1;             // row 0..127
  const int ah = t & 1;              // half 0/1

  v8f acc[2][4];
#pragma unroll
  for (int mi = 0; mi < 2; ++mi)
#pragma unroll
    for (int ni = 0; ni < 4; ++ni)
      acc[mi][ni] = v8f{};

  for (int kt = 0; kt < KTOT; kt += BK) {
    // ================= A tile: weights are contiguous in k for fixed co ====
    {
      const float* src = Wt + (size_t)(co0 + ar) * KTOT + kt + ah * 16;
      if (kt + BK < KTOT) __builtin_prefetch(src + BK, 0, 3); // global_prefetch_b8
      __bf16* adst = &As[ar * LDA + ah * 16];
#pragma unroll
      for (int v = 0; v < 2; ++v) {
        v4f f0 = *(const v4f*)(src + v * 8);
        v4f f1 = *(const v4f*)(src + v * 8 + 4);
        v8bf pk;
        pk[0] = (__bf16)f0[0]; pk[1] = (__bf16)f0[1];
        pk[2] = (__bf16)f0[2]; pk[3] = (__bf16)f0[3];
        pk[4] = (__bf16)f1[0]; pk[5] = (__bf16)f1[1];
        pk[6] = (__bf16)f1[2]; pk[7] = (__bf16)f1[3];
        *(v8bf*)(adst + v * 8) = pk;
      }
    }
    // ================= B tile: im2col gather with zero padding =============
    {
      __attribute__((aligned(16))) __bf16 tmp[16];
#pragma unroll
      for (int j = 0; j < 16; ++j) {
        const int k  = kt + kc * 16 + j;
        const int ci = k / 9;
        const int r  = k - ci * 9;
        const int kh = r / 3;
        const int kw = r - kh * 3;
        const int ih = oh + kh - 1;
        const int iw = ow + kw - 1;
        float val = 0.0f;
        if ((unsigned)ih < HW && (unsigned)iw < HW)
          val = X[(((size_t)nb * IN_C + ci) * HW + ih) * HW + iw];
        tmp[j] = (__bf16)val;
      }
      __bf16* bdst = &Bs[bn * LDB + kc * 16];
      *(v8bf*)(bdst)     = *(const v8bf*)(tmp);
      *(v8bf*)(bdst + 8) = *(const v8bf*)(tmp + 8);
    }

    __syncthreads();

    // ================= fragment loads (documented wave32 layouts) ==========
    // A 16x32 bf16: lanes 0-15 row M=l hold K {off..off+7, 16+off..}, off=0;
    //               lanes 16-31 same rows, off=8.
    const int aoff = (lane & 16) ? 8 : 0;
    Half16 af[2];
#pragma unroll
    for (int mi = 0; mi < 2; ++mi) {
      const __bf16* arow = &As[(wm * 32 + mi * 16 + l15) * LDA];
      af[mi].h[0] = *(const v8bf*)(arow + aoff);
      af[mi].h[1] = *(const v8bf*)(arow + 16 + aoff);
    }
    // B 32x16 bf16: lane holds column N=l15, K {off..off+7, 16+off..16+off+7};
    //               lanes 0-15 -> off=0, lanes 16-31 -> off=8.
    const int bko = (lane >> 4) * 8;
    Half16 bf[4];
#pragma unroll
    for (int ni = 0; ni < 4; ++ni) {
      const __bf16* brow = &Bs[(wn * 64 + ni * 16 + l15) * LDB + bko];
      bf[ni].h[0] = *(const v8bf*)(brow);
      bf[ni].h[1] = *(const v8bf*)(brow + 16);
    }

#pragma unroll
    for (int mi = 0; mi < 2; ++mi)
#pragma unroll
      for (int ni = 0; ni < 4; ++ni)
        acc[mi][ni] = __builtin_amdgcn_wmma_f32_16x16x32_bf16(
            false, af[mi].v, false, bf[ni].v,
            (short)0, acc[mi][ni], false, false);

    __syncthreads();
  }

  // ================= epilogue: D 16x16 f32 layout ==========================
  // lanes 0-15: N=lane, VGPR j -> M=j ; lanes 16-31: N=lane-16, VGPR j -> M=j+8
  const int mhalf = (lane & 16) ? 8 : 0;
#pragma unroll
  for (int mi = 0; mi < 2; ++mi) {
#pragma unroll
    for (int ni = 0; ni < 4; ++ni) {
      const int pe  = p0 + wn * 64 + ni * 16 + l15;
      const int nbe = pe / NPIX;
      const int ohw = pe - nbe * NPIX;
      const int mb  = co0 + wm * 32 + mi * 16 + mhalf;
#pragma unroll
      for (int j = 0; j < 8; ++j) {
        Out[((size_t)nbe * OUT_C + (mb + j)) * NPIX + ohw] = acc[mi][ni][j];
      }
    }
  }
}

extern "C" void kernel_launch(void* const* d_in, const int* in_sizes, int n_in,
                              void* d_out, int out_size, void* d_ws, size_t ws_size,
                              hipStream_t stream) {
  const float* X  = (const float*)d_in[0];   // 32*128*56*56 fp32
  const float* Wt = (const float*)d_in[1];   // 256*128*3*3  fp32
  float* Out = (float*)d_out;                // 32*256*56*56 fp32
  (void)in_sizes; (void)n_in; (void)out_size; (void)d_ws; (void)ws_size;

  dim3 grid(784, 2, 1);   // 100352/128 N-tiles x 256/128 M-tiles (exact)
  dim3 block(256, 1, 1);  // 8 wave32 waves
  conv_igemm_wmma_bf16<<<grid, block, 0, stream>>>(X, Wt, Out);
}
